// TFFunnelRelMultiheadAttention_42064909697349
// MI455X (gfx1250) — compile-verified, matrix operand order checked by reading, not run on hardware
//
#include <hip/hip_runtime.h>
#include <hip/hip_bf16.h>
#include <stdint.h>

// ---------------------------------------------------------------------------
// CDNA5 (gfx1250) bf16 WMMA implementation of TFFunnel relative attention.
// All five GEMM stages lower to v_wmma_f32_16x16x32_bf16; softmax/LN in f32.
// V is produced pre-transposed [B,NH,DH,S] so every WMMA operand load is a
// pair of 16-byte (b128) per-lane accesses.  No explicit prefetch: the
// working set (~70 MB) is GL2-resident and the compiler software-pipelines
// the clause-grouped b128 loads 2-3 K-slices ahead (partial s_wait_loadcnt),
// which hides GL2 latency better than prefetch instructions that break its
// scheduling.
// ---------------------------------------------------------------------------

typedef __bf16 bf16_t;
typedef __attribute__((ext_vector_type(16))) __bf16 v16bf;
typedef __attribute__((ext_vector_type(8)))  float  v8f;

#define Bv 4
#define Sv 1024
#define Dv 768
#define NHv 12
#define DHv 64

// K offset inside a 16-bit 16x32 A/B fragment for VGPR-pair v and lane-half hs
// (CDNA5 ISA §7.12.2): lanes 0-15 hold K {0..7,16..23}, lanes 16-31
// hold K {8..15,24..31}; each packed VGPR holds two consecutive K values.
__device__ __forceinline__ int kofs(int v, int hs) {
  return (v < 4 ? (v << 1) : 16 + ((v - 4) << 1)) + (hs << 3);
}

// Load a 16x32 bf16 fragment where this lane's 32 K-values are contiguous.
__device__ __forceinline__ v16bf load_frag(const bf16_t* p, int hs) {
  v16bf f;
#pragma unroll
  for (int v = 0; v < 8; ++v) {
    int k = kofs(v, hs);
    f[2 * v]     = p[k];
    f[2 * v + 1] = p[k + 1];
  }
  return f;
}

__device__ __forceinline__ v8f wmma_bf16(v16bf a, v16bf b, v8f c) {
  return __builtin_amdgcn_wmma_f32_16x16x32_bf16(false, a, false, b, (short)0, c,
                                                 false, false);
}

__device__ __forceinline__ v8f vzero8() {
  v8f z = {0.f, 0.f, 0.f, 0.f, 0.f, 0.f, 0.f, 0.f};
  return z;
}

// ---------------------------------------------------------------------------
// f32 -> bf16 flat cast with zero padding beyond n (for r padded to 2048 rows)
__global__ void k_cast(const float* __restrict__ src, bf16_t* __restrict__ dst,
                       int n, int total) {
  int i = blockIdx.x * 256 + threadIdx.x;
  if (i < total) dst[i] = (bf16_t)(i < n ? src[i] : 0.f);
}

// 768x768 f32 [K][N] -> bf16 [N][K] so B fragments read contiguous K pairs.
__global__ void k_transpose(const float* __restrict__ w, bf16_t* __restrict__ wt) {
  int i = blockIdx.x * 256 + threadIdx.x;
  if (i < Dv * Dv) {
    int k = i / Dv, n = i % Dv;
    wt[(size_t)n * Dv + k] = (bf16_t)w[i];
  }
}

// ---------------------------------------------------------------------------
// Projection GEMM: C[M x 768] = A[M x 768] * B[768 x 768]  (bf16 in, f32 acc)
// mode 0: Q -> three biased/scaled bf16 head-major outputs (q_w, q_r, q_s)
// mode 1:   K -> acc + bias, head-major [B,NH,S,DH]
// mode 2:   V -> acc + bias, TRANSPOSED [B,NH,DH,S] (for contiguous P@V reads)
// mode 3:   r_head -> [NH][2048][DH] bf16
__global__ void k_gemm_proj(const bf16_t* __restrict__ A, const bf16_t* __restrict__ BT,
                            bf16_t* out0, bf16_t* out1, bf16_t* out2,
                            const float* __restrict__ bias0,
                            const float* __restrict__ bias1,
                            const float* __restrict__ bias2, int mode) {
  const int lane = threadIdx.x & 31, w = threadIdx.x >> 5;
  const int hs = lane >> 4, ln = lane & 15;
  const int n0 = (blockIdx.x * 4 + w) * 16;
  const int m0 = blockIdx.y * 16;
  const bf16_t* arow = A + (size_t)(m0 + ln) * Dv;
  const bf16_t* brow = BT + (size_t)(n0 + ln) * Dv;
  v8f acc = vzero8();
#pragma unroll 4
  for (int ks = 0; ks < 24; ++ks) {
    v16bf a = load_frag(arow + ks * 32, hs);
    v16bf b = load_frag(brow + ks * 32, hs);
    acc = wmma_bf16(a, b, acc);
  }
  const int n = n0 + ln;
  const int h = n >> 6, d = n & 63;
#pragma unroll
  for (int r = 0; r < 8; ++r) {
    int row = m0 + r + 8 * hs;
    if (mode == 3) {
      out0[((size_t)h * 2048 + row) * DHv + d] = (bf16_t)acc[r];
    } else {
      int bb = row >> 10, s = row & 1023;
      if (mode == 0) {
        size_t dst = (((size_t)(bb * NHv + h)) * Sv + s) * DHv + d;
        out0[dst] = (bf16_t)((acc[r] + bias0[n]) * 0.125f);
        out1[dst] = (bf16_t)((acc[r] + bias1[n]) * 0.125f);
        out2[dst] = (bf16_t)((acc[r] + bias2[n]) * 0.125f);
      } else if (mode == 1) {
        size_t dst = (((size_t)(bb * NHv + h)) * Sv + s) * DHv + d;
        out0[dst] = (bf16_t)(acc[r] + bias0[n]);
      } else {  // mode 2: transposed V, contiguous in s per lane
        size_t dst = (((size_t)(bb * NHv + h)) * DHv + d) * Sv + s;
        out0[dst] = (bf16_t)(acc[r] + bias0[n]);
      }
    }
  }
}

// ---------------------------------------------------------------------------
// Attention: one block per (16-row i-tile, head, batch); 4 waves, 131 KB LDS.
__global__ void k_attention(const bf16_t* __restrict__ qw, const bf16_t* __restrict__ qr,
                            const bf16_t* __restrict__ qs, const bf16_t* __restrict__ kh,
                            const bf16_t* __restrict__ vT, const bf16_t* __restrict__ rh,
                            const float* __restrict__ seg_embed,
                            const unsigned char* __restrict__ ttmat,
                            const int* __restrict__ amask,
                            const float* __restrict__ cls_mask,
                            bf16_t* __restrict__ av) {
  extern __shared__ char smem[];
  float*  sc  = (float*)smem;                                   // 16 x 1024 scores
  float*  ps  = (float*)(smem + 16 * 1024 * 4);                 // 16 x 1040 pos strip
  float*  ttb = (float*)(smem + 16 * 1024 * 4 + 16 * 1040 * 4); // 2 x 16
  bf16_t* pb  = (bf16_t*)ps;  // reused after strip is consumed: 16x1024 bf16 probs

  const int lane = threadIdx.x & 31, w = threadIdx.x >> 5;
  const int hs = lane >> 4, ln = lane & 15;
  const int i0 = blockIdx.x * 16;
  const int h  = blockIdx.y;
  const int b  = blockIdx.z;
  const size_t bh = (size_t)b * NHv + h;
  const int tmin = 1009 - i0;  // 1024 - (i0 + 15)

  // Biased-query fragments (each wave keeps its own copy).
  const bf16_t* qwrow = qw + (bh * Sv + i0 + ln) * DHv;
  const bf16_t* qrrow = qr + (bh * Sv + i0 + ln) * DHv;
  v16bf qwf0 = load_frag(qwrow, hs),      qwf1 = load_frag(qwrow + 32, hs);
  v16bf qrf0 = load_frag(qrrow, hs),      qrf1 = load_frag(qrrow + 32, hs);

  // Token-type bias: (q_s . seg_embed[sel]) per row, wave 0 only.
  if (threadIdx.x < 32) {
    const bf16_t* q = qs + (bh * Sv + i0 + ln) * DHv;
    const float* sg = seg_embed + ((size_t)hs * NHv + h) * DHv;
    float s = 0.f;
#pragma unroll
    for (int d = 0; d < DHv; ++d) s += (float)q[d] * sg[d];
    ttb[hs * 16 + ln] = s;
  }

  // Positional strip: pos[i][t] = q_r[i] . r_head[t], t in [tmin, tmin+1038].
  for (int tt = w; tt < 65; tt += 4) {
    int t = tmin + tt * 16 + ln;
    t = t < 0 ? 0 : (t > 2046 ? 2046 : t);
    const bf16_t* rrow = rh + ((size_t)h * 2048 + t) * DHv;
    v8f acc = vzero8();
    acc = wmma_bf16(qrf0, load_frag(rrow, hs), acc);
    acc = wmma_bf16(qrf1, load_frag(rrow + 32, hs), acc);
#pragma unroll
    for (int r = 0; r < 8; ++r) ps[(r + 8 * hs) * 1040 + tt * 16 + ln] = acc[r];
  }

  // Content scores: q_w . k^T
  for (int jt = w; jt < 64; jt += 4) {
    int j0 = jt * 16;
    const bf16_t* krow = kh + (bh * Sv + j0 + ln) * DHv;
    v8f acc = vzero8();
    acc = wmma_bf16(qwf0, load_frag(krow, hs), acc);
    acc = wmma_bf16(qwf1, load_frag(krow + 32, hs), acc);
#pragma unroll
    for (int r = 0; r < 8; ++r) sc[(r + 8 * hs) * 1024 + j0 + ln] = acc[r];
  }
  __syncthreads();

  // Combine: content + diag-shifted positional*cls + token-type*cls + mask.
  for (int idx = threadIdx.x; idx < 16 * 1024; idx += 128) {
    int i = idx >> 10, j = idx & 1023;
    float cls = cls_mask[(size_t)(i0 + i) * Sv + j];
    float pos = ps[i * 1040 + (j - i + 15)];
    float tb = ttmat[((size_t)b * Sv + i0 + i) * Sv + j] ? ttb[16 + i] : ttb[i];
    float am = (float)amask[b * Sv + j];
    sc[idx] += (pos + tb) * cls - 1000000.0f * (1.0f - am);
  }
  __syncthreads();

  // Softmax: 8 lanes per row, shfl_xor reductions within 8-lane groups.
  {
    int row = threadIdx.x >> 3, sub = threadIdx.x & 7;
    float* srow = sc + row * 1024;
    float mx = -3.4e38f;
    for (int j = sub; j < 1024; j += 8) mx = fmaxf(mx, srow[j]);
#pragma unroll
    for (int o = 1; o < 8; o <<= 1) mx = fmaxf(mx, __shfl_xor(mx, o));
    float sum = 0.f;
    for (int j = sub; j < 1024; j += 8) {
      float e = __expf(srow[j] - mx);
      srow[j] = e;
      sum += e;
    }
#pragma unroll
    for (int o = 1; o < 8; o <<= 1) sum += __shfl_xor(sum, o);
    float inv = 1.f / sum;
    for (int j = sub; j < 1024; j += 8)
      pb[row * 1024 + j] = (bf16_t)(srow[j] * inv);  // strip LDS reused as bf16
  }
  __syncthreads();

  // P @ V: wave w owns d-tile w (DH = 4 x 16). V is [B,NH,DH,S] so the B
  // fragment reads contiguous K (sequence) pairs per lane -> b128 loads.
  {
    int d0 = w * 16;
    const bf16_t* vcol = vT + (bh * DHv + d0 + ln) * Sv;
    v8f acc = vzero8();
    for (int ks = 0; ks < 32; ++ks) {
      int j0 = ks * 32;
      v16bf a = load_frag(pb + ln * 1024 + j0, hs);
      v16bf bfr = load_frag(vcol + j0, hs);
      acc = wmma_bf16(a, bfr, acc);
    }
#pragma unroll
    for (int r = 0; r < 8; ++r) {
      int M = r + 8 * hs;
      av[((size_t)(b * Sv + i0 + M)) * Dv + h * DHv + d0 + ln] = (bf16_t)acc[r];
    }
  }
}

// ---------------------------------------------------------------------------
// Post projection + residual + LayerNorm.  Block = 16-row strip, 4 waves.
__global__ void k_post(const bf16_t* __restrict__ av, const bf16_t* __restrict__ WpT,
                       const float* __restrict__ bpost, const float* __restrict__ query,
                       const float* __restrict__ ln_g, const float* __restrict__ ln_b,
                       float* __restrict__ out) {
  __shared__ float xbuf[16 * Dv];  // 48 KB
  const int lane = threadIdx.x & 31, w = threadIdx.x >> 5;
  const int hs = lane >> 4, ln = lane & 15;
  const int m0 = blockIdx.x * 16;

  v8f acc[12];
#pragma unroll
  for (int t = 0; t < 12; ++t) acc[t] = vzero8();

  const bf16_t* arow = av + (size_t)(m0 + ln) * Dv;
  for (int ks = 0; ks < 24; ++ks) {
    v16bf a = load_frag(arow + ks * 32, hs);
#pragma unroll
    for (int t = 0; t < 12; ++t) {
      int n0 = (w * 12 + t) * 16;
      v16bf b = load_frag(WpT + (size_t)(n0 + ln) * Dv + ks * 32, hs);
      acc[t] = wmma_bf16(a, b, acc[t]);
    }
  }
#pragma unroll
  for (int t = 0; t < 12; ++t) {
    int col = (w * 12 + t) * 16 + ln;
#pragma unroll
    for (int r = 0; r < 8; ++r) {
      int row = r + 8 * hs;
      xbuf[row * Dv + col] =
          acc[t][r] + bpost[col] + query[(size_t)(m0 + row) * Dv + col];
    }
  }
  __syncthreads();

  int row = threadIdx.x >> 3, sub = threadIdx.x & 7;
  float s = 0.f, s2 = 0.f;
  for (int j = sub; j < Dv; j += 8) {
    float x = xbuf[row * Dv + j];
    s += x;
    s2 += x * x;
  }
#pragma unroll
  for (int o = 1; o < 8; o <<= 1) {
    s += __shfl_xor(s, o);
    s2 += __shfl_xor(s2, o);
  }
  float mean = s * (1.f / Dv);
  float var = s2 * (1.f / Dv) - mean * mean;
  float inv = rsqrtf(var + 1e-9f);
  for (int j = sub; j < Dv; j += 8)
    out[(size_t)(m0 + row) * Dv + j] = (xbuf[row * Dv + j] - mean) * inv * ln_g[j] + ln_b[j];
}

// ---------------------------------------------------------------------------
extern "C" void kernel_launch(void* const* d_in, const int* in_sizes, int n_in,
                              void* d_out, int out_size, void* d_ws, size_t ws_size,
                              hipStream_t stream) {
  const float* query  = (const float*)d_in[0];
  const float* key    = (const float*)d_in[1];
  const float* value  = (const float*)d_in[2];
  const float* r      = (const float*)d_in[3];
  const unsigned char* ttmat = (const unsigned char*)d_in[4];
  const int*   amask  = (const int*)d_in[5];
  const float* clsm   = (const float*)d_in[6];
  const float* Wq     = (const float*)d_in[7];
  const float* Wk     = (const float*)d_in[8];
  const float* bk     = (const float*)d_in[9];
  const float* Wv     = (const float*)d_in[10];
  const float* bv     = (const float*)d_in[11];
  const float* Wpost  = (const float*)d_in[12];
  const float* bpost  = (const float*)d_in[13];
  const float* ln_g   = (const float*)d_in[14];
  const float* ln_b   = (const float*)d_in[15];
  const float* r_w_b  = (const float*)d_in[16];
  const float* r_r_b  = (const float*)d_in[17];
  const float* r_ker  = (const float*)d_in[18];
  const float* r_s_b  = (const float*)d_in[19];
  const float* seg    = (const float*)d_in[20];
  float* out = (float*)d_out;

  char* ws = (char*)d_ws;
  size_t off = 0;
  auto alloc = [&](size_t bytes) -> void* {
    void* p = ws + off;
    off = (off + bytes + 255) & ~(size_t)255;
    return p;
  };
  const size_t MD = (size_t)Bv * Sv;        // 4096
  bf16_t* qbf   = (bf16_t*)alloc(MD * Dv * 2);
  bf16_t* kbf   = (bf16_t*)alloc(MD * Dv * 2);
  bf16_t* vbf   = (bf16_t*)alloc(MD * Dv * 2);
  bf16_t* rbf   = (bf16_t*)alloc((size_t)2048 * Dv * 2);
  bf16_t* WqT   = (bf16_t*)alloc((size_t)Dv * Dv * 2);
  bf16_t* WkT   = (bf16_t*)alloc((size_t)Dv * Dv * 2);
  bf16_t* WvT   = (bf16_t*)alloc((size_t)Dv * Dv * 2);
  bf16_t* WpT   = (bf16_t*)alloc((size_t)Dv * Dv * 2);
  bf16_t* rkT   = (bf16_t*)alloc((size_t)Dv * Dv * 2);
  bf16_t* q_w   = (bf16_t*)alloc(MD * Dv * 2);
  bf16_t* q_r   = (bf16_t*)alloc(MD * Dv * 2);
  bf16_t* q_s   = (bf16_t*)alloc(MD * Dv * 2);
  bf16_t* khd   = (bf16_t*)alloc(MD * Dv * 2);
  bf16_t* vTd   = (bf16_t*)alloc(MD * Dv * 2);   // [B,NH,DH,S]
  bf16_t* rhd   = (bf16_t*)alloc((size_t)NHv * 2048 * DHv * 2);
  bf16_t* avec  = (bf16_t*)alloc(MD * Dv * 2);
  (void)ws_size; (void)n_in; (void)in_sizes; (void)out_size;

  const int NQ = (int)(MD * Dv);
  k_cast<<<(NQ + 255) / 256, 256, 0, stream>>>(query, qbf, NQ, NQ);
  k_cast<<<(NQ + 255) / 256, 256, 0, stream>>>(key, kbf, NQ, NQ);
  k_cast<<<(NQ + 255) / 256, 256, 0, stream>>>(value, vbf, NQ, NQ);
  const int NRs = 2047 * Dv, NRt = 2048 * Dv;
  k_cast<<<(NRt + 255) / 256, 256, 0, stream>>>(r, rbf, NRs, NRt);

  const int NW = Dv * Dv;
  k_transpose<<<(NW + 255) / 256, 256, 0, stream>>>(Wq, WqT);
  k_transpose<<<(NW + 255) / 256, 256, 0, stream>>>(Wk, WkT);
  k_transpose<<<(NW + 255) / 256, 256, 0, stream>>>(Wv, WvT);
  k_transpose<<<(NW + 255) / 256, 256, 0, stream>>>(Wpost, WpT);
  k_transpose<<<(NW + 255) / 256, 256, 0, stream>>>(r_ker, rkT);

  // Projections: grid.x covers 48 n-tiles / 4 waves, grid.y covers m-tiles.
  k_gemm_proj<<<dim3(12, 256), 128, 0, stream>>>(qbf, WqT, q_w, q_r, q_s,
                                                 r_w_b, r_r_b, r_s_b, 0);
  k_gemm_proj<<<dim3(12, 256), 128, 0, stream>>>(kbf, WkT, khd, nullptr, nullptr,
                                                 bk, nullptr, nullptr, 1);
  k_gemm_proj<<<dim3(12, 256), 128, 0, stream>>>(vbf, WvT, vTd, nullptr, nullptr,
                                                 bv, nullptr, nullptr, 2);
  k_gemm_proj<<<dim3(12, 128), 128, 0, stream>>>(rbf, rkT, rhd, nullptr, nullptr,
                                                 nullptr, nullptr, nullptr, 3);

  const size_t smem = 16 * 1024 * 4 + 16 * 1040 * 4 + 2 * 16 * 4;  // 132224 B
  k_attention<<<dim3(Sv / 16, NHv, Bv), 128, smem, stream>>>(
      q_w, q_r, q_s, khd, vTd, rhd, seg, ttmat, amask, clsm, avec);

  k_post<<<MD / 16, 128, 0, stream>>>(avec, WpT, bpost, query, ln_g, ln_b, out);
}
